// FastAttention_2345052143998
// MI455X (gfx1250) — matrix operand order repaired and checked by reading, not compiled, not run on hardware
//
#include <hip/hip_runtime.h>
#include <hip/hip_bf16.h>
#include <math.h>

typedef __attribute__((ext_vector_type(16))) _Float16 v16h;
typedef __attribute__((ext_vector_type(8)))  _Float16 v8h;
typedef __attribute__((ext_vector_type(4)))  _Float16 v4h;
typedef __attribute__((ext_vector_type(8)))  float    v8f;

#define BH_TOT  64
#define SEQLEN  4096
#define DHEAD   64
#define NBF     256
#define DATA_NORM 0.35355339059327379f   // 64^-0.25
#define RATIO_M   0.0625f                // 256^-0.5
#define KEPS_C    1e-4f

// ---- WMMA fragment loads from LDS, vectorized to ds_load_b128 ----
// A (16x32 f16) from row-major [M][K]: lane m = lane%16; per-lane halves are two
// contiguous 8-half runs at K = kbase+kh and K = kbase+kh+16 (kh = 8*(lane/16)).
static __device__ __forceinline__ v16h lds_load_a16(const _Float16* base, int lda, int kbase, int lane) {
  const int m  = lane & 15;
  const int kh = (lane >> 4) << 3;
  const _Float16* p = base + m * lda + kbase + kh;
  v8h lo = *(const v8h*)p;
  v8h hi = *(const v8h*)(p + 16);
  return __builtin_shufflevector(lo, hi, 0,1,2,3,4,5,6,7,8,9,10,11,12,13,14,15);
}
// B (32x16 f16) from N-major storage Bt[N][Kext]: lane n = n0+lane%16; per-lane
// halves are two contiguous 8-half runs at K = kbase+kh and K = kbase+kh+16 (kh = 8*(lane/16)).
static __device__ __forceinline__ v16h lds_load_bt16(const _Float16* base, int ldk, int kbase, int n0, int lane) {
  const int n  = n0 + (lane & 15);
  const int kh = (lane >> 4) << 3;
  const _Float16* p = base + n * ldk + kbase + kh;
  v8h lo = *(const v8h*)p;
  v8h hi = *(const v8h*)(p + 16);
  return __builtin_shufflevector(lo, hi, 0,1,2,3,4,5,6,7,8,9,10,11,12,13,14,15);
}
static __device__ __forceinline__ v8f wmma_f16(v16h a, v16h b, v8f c) {
  return __builtin_amdgcn_wmma_f32_16x16x32_f16(false, a, false, b, (short)0, c, false, false);
}
static __device__ __forceinline__ v4h cvt4(float4 f, float s) {
  v4h h;
  h[0] = (_Float16)(f.x * s); h[1] = (_Float16)(f.y * s);
  h[2] = (_Float16)(f.z * s); h[3] = (_Float16)(f.w * s);
  return h;
}

// ================= Pass 1: per-(b,h) global max of raw K-dash =================
// grid (256, 64), block 256 (8 waves); block: 16 seq rows x all 256 features
__global__ void kdash_max_kernel(const float* __restrict__ kin,
                                 const float* __restrict__ proj,
                                 float* __restrict__ blockmax) {
  __shared__ __align__(32) _Float16 Ak[16][DHEAD];
  __shared__ __align__(32) _Float16 Bpt[NBF][DHEAD];   // proj native [m][d] == N-major B
  __shared__ float red[256];
  const int tid = threadIdx.x, lane = tid & 31, wave = tid >> 5;
  const int bh = blockIdx.y, n0 = blockIdx.x << 4;
  {
    const size_t base = ((size_t)bh * SEQLEN + n0) * DHEAD;
    const int e = tid << 2;                            // 1024 elems, 4/thread
    float4 f = *(const float4*)&kin[base + e];
    *(v4h*)&Ak[e >> 6][e & 63] = cvt4(f, DATA_NORM);
  }
#pragma unroll
  for (int u = 0; u < 16; ++u) {
    const int e = (tid << 2) + (u << 10);              // 16384 elems flat copy
    float4 f = *(const float4*)&proj[e];
    *(v4h*)&(&Bpt[0][0])[e] = cvt4(f, 1.0f);
  }
  __syncthreads();
  float lmax = -1e30f;
  v16h a0 = lds_load_a16(&Ak[0][0], DHEAD, 0, lane);
  v16h a1 = lds_load_a16(&Ak[0][0], DHEAD, 32, lane);
#pragma unroll
  for (int tt = 0; tt < 2; ++tt) {
    const int ct = wave + (tt << 3);                   // feature tile 0..15
    v8f c = {};
    c = wmma_f16(a0, lds_load_bt16(&Bpt[0][0], DHEAD, 0,  ct << 4, lane), c);
    c = wmma_f16(a1, lds_load_bt16(&Bpt[0][0], DHEAD, 32, ct << 4, lane), c);
#pragma unroll
    for (int vv = 0; vv < 8; ++vv) lmax = fmaxf(lmax, c[vv]);
  }
  red[tid] = lmax;
  __syncthreads();
  for (int s = 128; s > 0; s >>= 1) {
    if (tid < s) red[tid] = fmaxf(red[tid], red[tid + s]);
    __syncthreads();
  }
  if (tid == 0) blockmax[bh * gridDim.x + blockIdx.x] = red[0];
}

__global__ void reduce_kmax_kernel(const float* __restrict__ blockmax,
                                   float* __restrict__ kmax) {
  __shared__ float red[256];
  const int bh = blockIdx.x, tid = threadIdx.x;
  red[tid] = blockmax[bh * 256 + tid];
  __syncthreads();
  for (int s = 128; s > 0; s >>= 1) {
    if (tid < s) red[tid] = fmaxf(red[tid], red[tid + s]);
    __syncthreads();
  }
  if (tid == 0) kmax[bh] = red[0];
}

// ===== Pass 2: context^T = (k'^T v)^T and k_cumsum; k' recomputed, never spilled =====
// grid (16, 64), block 128 (4 waves); block owns one 16-feature tile, streams all n
__global__ void context_kernel(const float* __restrict__ kin,
                               const float* __restrict__ vin,
                               const float* __restrict__ proj,
                               const float* __restrict__ kmax,
                               float* __restrict__ context_t,   // [bh][e][m]
                               float* __restrict__ kcum) {
  __shared__ __align__(32) _Float16 Ak[64][DHEAD];     // k chunk, row-major (A)
  __shared__ __align__(32) _Float16 Bvt[DHEAD][64];    // v chunk, e-major (B^T)
  __shared__ __align__(32) _Float16 Bpt[16][DHEAD];    // proj tile, native (B^T)
  __shared__ __align__(32) _Float16 Akt[16][64];       // k' [feature][chunk-n] (A)
  __shared__ float sdiag[64];
  const int tid = threadIdx.x, lane = tid & 31, wave = tid >> 5;
  const int bh = blockIdx.y, m0 = blockIdx.x << 4;
  const float stab = kmax[bh];
#pragma unroll
  for (int u = 0; u < 2; ++u) {
    const int e = (tid << 2) + (u << 9);               // 1024 elems flat
    float4 f = *(const float4*)&proj[(size_t)m0 * DHEAD + e];
    *(v4h*)&(&Bpt[0][0])[e] = cvt4(f, 1.0f);
  }
  v8f cctx = {};
  float ksum = 0.f;
  const size_t kvbase = (size_t)bh * SEQLEN * DHEAD;
  for (int nc = 0; nc < SEQLEN; nc += 64) {
    __syncthreads();                                   // prev iter done with LDS
#pragma unroll
    for (int u = 0; u < 8; ++u) {
      const int e = (tid << 2) + (u << 9);             // 4096 elems, 4-contig/thread
      const int r = e >> 6, cc = e & 63;
      float4 fk = *(const float4*)&kin[kvbase + (size_t)nc * DHEAD + e];
      *(v4h*)&Ak[r][cc] = cvt4(fk, DATA_NORM);
      float4 fv = *(const float4*)&vin[kvbase + (size_t)nc * DHEAD + e];
      Bvt[cc + 0][r] = (_Float16)fv.x;                 // transpose into e-major
      Bvt[cc + 1][r] = (_Float16)fv.y;
      Bvt[cc + 2][r] = (_Float16)fv.z;
      Bvt[cc + 3][r] = (_Float16)fv.w;
    }
    if (nc + 64 < SEQLEN) {
      __builtin_prefetch(&kin[kvbase + (size_t)(nc + 64) * DHEAD + tid * 8], 0, 0);
      __builtin_prefetch(&vin[kvbase + (size_t)(nc + 64) * DHEAD + tid * 8], 0, 0);
    }
    __syncthreads();
    if (tid < 64) {
      float s = 0.f;
#pragma unroll
      for (int d = 0; d < 64; ++d) { float x = (float)Ak[tid][d]; s += x * x; }
      sdiag[tid] = 0.5f * s;
    }
    // dash for this wave's 16-row sub-chunk
    v8f cd = {};
    {
      const _Float16* arow = &Ak[wave << 4][0];
      cd = wmma_f16(lds_load_a16(arow, DHEAD, 0,  lane),
                    lds_load_bt16(&Bpt[0][0], DHEAD, 0,  0, lane), cd);
      cd = wmma_f16(lds_load_a16(arow, DHEAD, 32, lane),
                    lds_load_bt16(&Bpt[0][0], DHEAD, 32, 0, lane), cd);
    }
    __syncthreads();                                   // sdiag visible
    {
      const int row = (wave << 4) + ((lane >> 4) << 3);
      const int col = lane & 15;
#pragma unroll
      for (int vv = 0; vv < 8; ++vv) {
        float kp = RATIO_M * (__expf(cd[vv] - sdiag[row + vv] - stab) + KEPS_C);
        Akt[col][row + vv] = (_Float16)kp;
      }
    }
    __syncthreads();                                   // Akt complete
    if (tid < 16) {
      float s = 0.f;
#pragma unroll
      for (int j = 0; j < 64; ++j) s += (float)Akt[tid][j];
      ksum += s;
    }
    cctx = wmma_f16(lds_load_a16(&Akt[0][0], 64, 0,  lane),
                    lds_load_bt16(&Bvt[0][0], 64, 0,  wave << 4, lane), cctx);
    cctx = wmma_f16(lds_load_a16(&Akt[0][0], 64, 32, lane),
                    lds_load_bt16(&Bvt[0][0], 64, 32, wave << 4, lane), cctx);
  }
  {
    const int row = (lane >> 4) << 3, col = lane & 15;
#pragma unroll
    for (int vv = 0; vv < 8; ++vv)                     // store transposed: [e][m]
      context_t[((size_t)bh * DHEAD + (wave << 4) + col) * NBF + m0 + row + vv] = cctx[vv];
  }
  if (tid < 16) kcum[bh * NBF + m0 + tid] = ksum;
}

// ======== Pass 3: out = (q' @ context) * D_inv, q' recomputed in LDS ==========
// grid (256, 64), block 128 (4 waves); block: 16 seq rows x all 64 out dims
__global__ void output_kernel(const float* __restrict__ qin,
                              const float* __restrict__ proj,
                              const float* __restrict__ context_t,
                              const float* __restrict__ kcum,
                              float* __restrict__ out) {
  __shared__ __align__(32) _Float16 Aq[16][DHEAD];
  __shared__ __align__(32) _Float16 Bpt[NBF][DHEAD];   // proj native (B^T for dash)
  __shared__ __align__(32) float    Dash[16][NBF];
  __shared__ __align__(32) _Float16 Qp[16][NBF];       // q' row-major (A)
  __shared__ __align__(32) _Float16 Bct[DHEAD][NBF];   // context_t native (B^T for out)
  __shared__ float kc[NBF];
  __shared__ float sdiag[16], rmax[16], dinv[16];
  const int tid = threadIdx.x, lane = tid & 31, wave = tid >> 5;
  const int bh = blockIdx.y, n0 = blockIdx.x << 4;
#pragma unroll
  for (int u = 0; u < 2; ++u) {
    const int e = (tid << 2) + (u << 9);
    float4 f = *(const float4*)&qin[((size_t)bh * SEQLEN + n0) * DHEAD + e];
    *(v4h*)&Aq[e >> 6][e & 63] = cvt4(f, DATA_NORM);
  }
#pragma unroll
  for (int u = 0; u < 32; ++u) {
    const int e = (tid << 2) + (u << 9);               // 16384 elems flat copies
    float4 fp = *(const float4*)&proj[e];
    *(v4h*)&(&Bpt[0][0])[e] = cvt4(fp, 1.0f);
    float4 fc = *(const float4*)&context_t[(size_t)bh * NBF * DHEAD + e];
    *(v4h*)&(&Bct[0][0])[e] = cvt4(fc, 1.0f);
  }
#pragma unroll
  for (int u = 0; u < 2; ++u) {
    const int e = tid + (u << 7);
    kc[e] = kcum[bh * NBF + e];
  }
  __syncthreads();
  // q dash: 4 feature tiles per wave -> Dash
  {
    v16h a0 = lds_load_a16(&Aq[0][0], DHEAD, 0,  lane);
    v16h a1 = lds_load_a16(&Aq[0][0], DHEAD, 32, lane);
#pragma unroll
    for (int t = 0; t < 4; ++t) {
      const int ct = wave * 4 + t;
      v8f c = {};
      c = wmma_f16(a0, lds_load_bt16(&Bpt[0][0], DHEAD, 0,  ct << 4, lane), c);
      c = wmma_f16(a1, lds_load_bt16(&Bpt[0][0], DHEAD, 32, ct << 4, lane), c);
      const int row = (lane >> 4) << 3, col = (ct << 4) + (lane & 15);
#pragma unroll
      for (int vv = 0; vv < 8; ++vv) Dash[row + vv][col] = c[vv];
    }
  }
  __syncthreads();
  if (tid < 16) {
    float s = 0.f;
#pragma unroll
    for (int d = 0; d < 64; ++d) { float x = (float)Aq[tid][d]; s += x * x; }
    sdiag[tid] = 0.5f * s;
    float m = -1e30f;
    const float4* dp = (const float4*)&Dash[tid][0];
#pragma unroll
    for (int j = 0; j < NBF / 4; ++j) {
      float4 dd = dp[j];
      m = fmaxf(m, fmaxf(fmaxf(dd.x, dd.y), fmaxf(dd.z, dd.w)));
    }
    rmax[tid] = m;
  }
  __syncthreads();
#pragma unroll
  for (int u = 0; u < 32; ++u) {
    const int e = (tid << 5) + u;                      // 32-contig halves per thread
    const int r = e >> 8, cc = e & 255;
    Qp[r][cc] = (_Float16)(RATIO_M * (__expf(Dash[r][cc] - sdiag[r] - rmax[r]) + KEPS_C));
  }
  __syncthreads();
  if (tid < 16) {
    float denom = 0.f;
    for (int m = 0; m < NBF; ++m) denom += (float)Qp[tid][m] * kc[m];
    dinv[tid] = 1.0f / denom;
  }
  __syncthreads();
  // out tile: this wave's 16-col slice of e, K = 256 features
  v8f c = {};
#pragma unroll
  for (int ks = 0; ks < NBF; ks += 32) {
    c = wmma_f16(lds_load_a16(&Qp[0][0], NBF, ks, lane),
                 lds_load_bt16(&Bct[0][0], NBF, ks, wave << 4, lane), c);
  }
  {
    const int row = (lane >> 4) << 3, col = lane & 15;
#pragma unroll
    for (int vv = 0; vv < 8; ++vv) {
      const int r = row + vv;
      out[((size_t)bh * SEQLEN + n0 + r) * DHEAD + (wave << 4) + col] = c[vv] * dinv[r];
    }
  }
}

extern "C" void kernel_launch(void* const* d_in, const int* in_sizes, int n_in,
                              void* d_out, int out_size, void* d_ws, size_t ws_size,
                              hipStream_t stream) {
  (void)in_sizes; (void)n_in; (void)out_size; (void)ws_size;
  const float* q    = (const float*)d_in[0];
  const float* k    = (const float*)d_in[1];
  const float* v    = (const float*)d_in[2];
  const float* proj = (const float*)d_in[3];
  float* out = (float*)d_out;
  // workspace: blockmax[64*256] | kmax[64] | kcum[64*256] | context_t[64*64*256]  (~4.2 MB)
  float* blockmax  = (float*)d_ws;
  float* kmax      = blockmax + BH_TOT * 256;
  float* kcum      = kmax + BH_TOT;
  float* context_t = kcum + BH_TOT * NBF;

  kdash_max_kernel  <<<dim3(SEQLEN / 16, BH_TOT), 256, 0, stream>>>(k, proj, blockmax);
  reduce_kmax_kernel<<<dim3(BH_TOT),              256, 0, stream>>>(blockmax, kmax);
  context_kernel    <<<dim3(NBF / 16, BH_TOT),    128, 0, stream>>>(k, v, proj, kmax, context_t, kcum);
  output_kernel     <<<dim3(SEQLEN / 16, BH_TOT), 128, 0, stream>>>(q, proj, context_t, kcum, out);
}
